// VectorQuantize_20667382629200
// MI455X (gfx1250) — matrix operand order repaired and checked by reading, not compiled, not run on hardware
//
#include <hip/hip_runtime.h>
#include <hip/hip_bf16.h>

typedef __attribute__((ext_vector_type(16))) __bf16 v16bf;
typedef __attribute__((ext_vector_type(8)))  __bf16 v8bf;
typedef __attribute__((ext_vector_type(4)))  __bf16 v4bf;
typedef __attribute__((ext_vector_type(8)))  float  v8f;

#define D_DIM  1024
#define CD_DIM 256
#define K_DIM  8192
#define T_DIM  16384

// ---- WMMA fragment loaders (bf16, 16x16x32) -------------------------------
// A 16x32 (M x K), per ISA: lane half h = l/16, M = l%16,
//   a[i], i<8  -> K = h*8 + i ; i>=8 -> K = 16 + h*8 + (i-8)
__device__ __forceinline__ v16bf ld_frag_a(const __bf16* base, int ldk, int m, int half) {
  const v8bf lo = *(const v8bf*)(base + (size_t)m * ldk + half * 8);
  const v8bf hi = *(const v8bf*)(base + (size_t)m * ldk + 16 + half * 8);
  return __builtin_shufflevector(lo, hi, 0,1,2,3,4,5,6,7,8,9,10,11,12,13,14,15);
}
// B 32x16 (K x N), same K order as A:
//   b[i], i<8 -> K = half*8 + i ; i>=8 -> K = 16 + half*8 + (i-8). Stored n-major in LDS: B[n][k].
__device__ __forceinline__ v16bf ld_frag_b(const __bf16* base, int ldk, int n, int half) {
  const v8bf lo = *(const v8bf*)(base + (size_t)n * ldk + half * 8);
  const v8bf hi = *(const v8bf*)(base + (size_t)n * ldk + 16 + half * 8);
  return __builtin_shufflevector(lo, hi, 0,1,2,3,4,5,6,7,8,9,10,11,12,13,14,15);
}
__device__ __forceinline__ v8f wmma_bf16(v16bf a, v16bf b, v8f c) {
  return __builtin_amdgcn_wmma_f32_16x16x32_bf16(false, a, false, b, (short)0, c, false, false);
}
__device__ __forceinline__ v8f zero8() { v8f z = {0,0,0,0,0,0,0,0}; return z; }

// ---- prep: row-wise weight-norm (optionally clipped unit-norm) -> bf16 ----
__global__ __launch_bounds__(256) void k_rownorm(const float* __restrict__ V,
                                                 const float* __restrict__ g,
                                                 __bf16* __restrict__ W,
                                                 int C, int useClip) {
  int row = blockIdx.x;
  const float* vr = V + (size_t)row * C;
  float s = 0.f;
  for (int i = threadIdx.x; i < C; i += 256) { float x = vr[i]; s += x * x; }
  for (int o = 16; o > 0; o >>= 1) s += __shfl_xor(s, o, 32);
  __shared__ float red[8];
  __shared__ float sc_s;
  int lane = threadIdx.x & 31, w = threadIdx.x >> 5;
  if (lane == 0) red[w] = s;
  __syncthreads();
  if (threadIdx.x == 0) {
    float tot = 0.f;
    for (int i = 0; i < 8; i++) tot += red[i];
    float nv = sqrtf(tot);
    sc_s = useClip ? (1.0f / fmaxf(nv, 1e-12f)) : ((g ? g[row] : 1.0f) / nv);
  }
  __syncthreads();
  float sc = sc_s;
  __bf16* wr = W + (size_t)row * C;
  for (int i = threadIdx.x; i < C; i += 256) wr[i] = (__bf16)(vr[i] * sc);
}

// ---- prep: f32 -> bf16 bulk convert (float4 -> 4x bf16) -------------------
__global__ void k_f32_to_bf16(const float* __restrict__ x, __bf16* __restrict__ y,
                              long long n4) {
  long long i = (long long)blockIdx.x * blockDim.x + threadIdx.x;
  if (i >= n4) return;
  float4 v = ((const float4*)x)[i];
  v4bf o = {(__bf16)v.x, (__bf16)v.y, (__bf16)v.z, (__bf16)v.w};
  ((v4bf*)y)[i] = o;
}

// ---- kernel 1: z_e = Win @ z + b, fused per-token L2 normalize -> enc -----
// Block: full M=256 strip x 32 tokens, K=1024 in 32-chunks. 8 waves, 4 tiles/wave.
__global__ __launch_bounds__(256) void k_inproj_enc(const __bf16* __restrict__ Win,
                                                    const __bf16* __restrict__ zb,
                                                    const float* __restrict__ in_b,
                                                    __bf16* __restrict__ enc) {
  __shared__ __align__(16) char smem[33 * 1024];
  __bf16* Asm = (__bf16*)smem;              // 256 x 32 (16KB), loop phase
  __bf16* Bsm = (__bf16*)(smem + 16384);    // 32 x 32 (2KB),  loop phase
  float* outsm = (float*)smem;              // 32 x 256 (32KB), epilogue (overlays A/B)
  float* nrm   = (float*)(smem + 32768);    // 32 floats
  int t = threadIdx.x, lane = t & 31, w = t >> 5, half = lane >> 4, col = lane & 15;
  int n0 = blockIdx.x * 32;

  v8f acc[4];
  acc[0] = zero8(); acc[1] = zero8(); acc[2] = zero8(); acc[3] = zero8();

  for (int k0 = 0; k0 < D_DIM; k0 += 32) {
    __syncthreads();
    { // stage A chunk: 256 rows x 32 k (64B per thread, coalesced)
      const uint4* s = (const uint4*)(Win + (size_t)t * D_DIM + k0);
      uint4* d = (uint4*)(Asm + t * 32);
      d[0] = s[0]; d[1] = s[1]; d[2] = s[2]; d[3] = s[3];
    }
    { // stage B chunk: Bsm[n][k] = zb[(k0+k)*T + n0+n]
      int k = t >> 3, nn = (t & 7) * 4;
      const __bf16* s = zb + (size_t)(k0 + k) * T_DIM + n0 + nn;
      Bsm[(nn + 0) * 32 + k] = s[0];
      Bsm[(nn + 1) * 32 + k] = s[1];
      Bsm[(nn + 2) * 32 + k] = s[2];
      Bsm[(nn + 3) * 32 + k] = s[3];
    }
    __syncthreads();
    v16bf b0 = ld_frag_b(Bsm, 32, col, half);
    v16bf b1 = ld_frag_b(Bsm, 32, 16 + col, half);
    v16bf a0 = ld_frag_a(Asm, 32, (2 * w + 0) * 16 + col, half);
    v16bf a1 = ld_frag_a(Asm, 32, (2 * w + 1) * 16 + col, half);
    acc[0] = wmma_bf16(a0, b0, acc[0]);
    acc[1] = wmma_bf16(a0, b1, acc[1]);
    acc[2] = wmma_bf16(a1, b0, acc[2]);
    acc[3] = wmma_bf16(a1, b1, acc[3]);
  }
  __syncthreads();
  // epilogue: bias, transpose into LDS [token][cd]
#pragma unroll
  for (int mt2 = 0; mt2 < 2; mt2++)
#pragma unroll
    for (int nt = 0; nt < 2; nt++)
#pragma unroll
      for (int r = 0; r < 8; r++) {
        int m = (2 * w + mt2) * 16 + r + 8 * half;
        int n = nt * 16 + col;
        outsm[n * 256 + m] = acc[mt2 * 2 + nt][r] + in_b[m];
      }
  __syncthreads();
  // per-token inverse norms: wave w owns tokens w*4..w*4+3
  for (int q = 0; q < 4; q++) {
    int n = w * 4 + q;
    float s = 0.f;
#pragma unroll
    for (int j = 0; j < 8; j++) { float v = outsm[n * 256 + lane + 32 * j]; s += v * v; }
    for (int o = 16; o > 0; o >>= 1) s += __shfl_xor(s, o, 32);
    if (lane == 0) nrm[n] = 1.0f / fmaxf(sqrtf(s), 1e-12f);
  }
  __syncthreads();
  // coalesced bf16 writes: enc[(token)][cd]
  for (int n = 0; n < 32; n++)
    enc[(size_t)(n0 + n) * CD_DIM + t] = (__bf16)(outsm[n * 256 + t] * nrm[n]);
}

// ---- kernel 2: scores = enc @ cbu^T, fused running argmax over 8192 codes -
// Block: 32 tokens; codes in 128-chunks (8 waves x 16), CD in 128-chunks.
// Per wave per code-chunk: 16 WMMAs / 4 barriers; B frags reused across 2 M-tiles.
__global__ __launch_bounds__(256) void k_argmax(const __bf16* __restrict__ enc,
                                                const __bf16* __restrict__ cbu,
                                                int* __restrict__ idx_out,
                                                float* __restrict__ fidx_out) {
  __shared__ __align__(16) char smem[16384 + 32768];
  __bf16* Asm = (__bf16*)smem;              // 32 x 256 (16KB)
  __bf16* Bsm = (__bf16*)(smem + 16384);    // 128 x 128 (32KB)
  int t = threadIdx.x, lane = t & 31, w = t >> 5, half = lane >> 4, col = lane & 15;
  int tok0 = blockIdx.x * 32;

  { // stage enc tile once (64B/thread, coalesced)
    int m = t >> 3, part = (t & 7) * 32;
    const uint4* s = (const uint4*)(enc + (size_t)(tok0 + m) * CD_DIM + part);
    uint4* d = (uint4*)(Asm + m * CD_DIM + part);
    d[0] = s[0]; d[1] = s[1]; d[2] = s[2]; d[3] = s[3];
  }

  float bestv[16];
  int besti[16];
#pragma unroll
  for (int r = 0; r < 16; r++) { bestv[r] = -3.0e38f; besti[r] = 0; }

  for (int c0 = 0; c0 < K_DIM; c0 += 128) {
    if (c0 + 128 < K_DIM)  // prefetch next codebook chunk (global_prefetch_b8)
      __builtin_prefetch(cbu + (size_t)(c0 + 128 + (t >> 1)) * CD_DIM, 0, 1);
    v8f acc0 = zero8(), acc1 = zero8();
#pragma unroll
    for (int kc = 0; kc < CD_DIM; kc += 128) {
      __syncthreads();
      { // stage B: 128 codes x 128 cd (128B/thread, coalesced rows)
        int c = t >> 1, part = (t & 1) * 64;
        const uint4* s = (const uint4*)(cbu + (size_t)(c0 + c) * CD_DIM + kc + part);
        uint4* d = (uint4*)(Bsm + c * 128 + part);
        d[0] = s[0]; d[1] = s[1]; d[2] = s[2]; d[3] = s[3];
        d[4] = s[4]; d[5] = s[5]; d[6] = s[6]; d[7] = s[7];
      }
      __syncthreads();
#pragma unroll
      for (int ks = 0; ks < 4; ks++) {
        v16bf b  = ld_frag_b(Bsm + ks * 32, 128, w * 16 + col, half);
        v16bf a0 = ld_frag_a(Asm + kc + ks * 32, CD_DIM, col, half);
        v16bf a1 = ld_frag_a(Asm + kc + ks * 32, CD_DIM, 16 + col, half);
        acc0 = wmma_bf16(a0, b, acc0);
        acc1 = wmma_bf16(a1, b, acc1);
      }
    }
    int code = c0 + w * 16 + col;
#pragma unroll
    for (int r = 0; r < 8; r++) {  // strict > keeps first (lowest-index) max
      float v0 = acc0[r];
      if (v0 > bestv[r]) { bestv[r] = v0; besti[r] = code; }
      float v1 = acc1[r];
      if (v1 > bestv[8 + r]) { bestv[8 + r] = v1; besti[8 + r] = code; }
    }
  }
  __syncthreads();
  // merge 128 per-column candidates per token (overlay LDS)
  float* bv = (float*)smem;               // 32 x 128 floats (16KB)
  int*   bi = (int*)(smem + 16384);       // 32 x 128 ints  (16KB)
  int e = w * 16 + col;
#pragma unroll
  for (int mt = 0; mt < 2; mt++)
#pragma unroll
    for (int r = 0; r < 8; r++) {
      int m = mt * 16 + r + 8 * half;
      bv[m * 128 + e] = bestv[mt * 8 + r];
      bi[m * 128 + e] = besti[mt * 8 + r];
    }
  __syncthreads();
  if (t < 32) {
    float mv = -3.0e38f; int mi = 0x7fffffff;
    for (int e2 = 0; e2 < 128; e2++) {
      float v = bv[t * 128 + e2]; int i2 = bi[t * 128 + e2];
      if (v > mv || (v == mv && i2 < mi)) { mv = v; mi = i2; }
    }
    idx_out[tok0 + t]  = mi;
    fidx_out[tok0 + t] = (float)mi;
  }
}

// ---- kernel 3: out = Wout @ codebook[idx] + out_b -------------------------
// Block: 128(M) x 32(N); gather z_q columns from codebook on the fly (f32->bf16).
__global__ __launch_bounds__(256) void k_outproj(const __bf16* __restrict__ Wout,
                                                 const float* __restrict__ codebook,
                                                 const int* __restrict__ idx,
                                                 const float* __restrict__ out_b,
                                                 float* __restrict__ out) {
  __shared__ __align__(16) char smem[8192 + 2048 + 256];
  __bf16* Asm = (__bf16*)smem;                   // 128 x 32
  __bf16* Bsm = (__bf16*)(smem + 8192);          // 32 x 32 (n-major)
  int* idxs   = (int*)(smem + 8192 + 2048);      // 32
  int t = threadIdx.x, lane = t & 31, w = t >> 5, half = lane >> 4, col = lane & 15;
  int n0 = blockIdx.x * 32;
  int m0 = blockIdx.y * 128;
  if (t < 32) idxs[t] = idx[n0 + t];

  v8f acc0 = zero8(), acc1 = zero8();
  for (int k0 = 0; k0 < CD_DIM; k0 += 32) {
    __syncthreads();
    { // stage A: Wout rows (32B/thread)
      int row = t >> 1, part = (t & 1) * 16;
      const uint4* s = (const uint4*)(Wout + (size_t)(m0 + row) * CD_DIM + k0 + part);
      uint4* d = (uint4*)(Asm + row * 32 + part);
      d[0] = s[0]; d[1] = s[1];
    }
    { // stage B: gather codebook row per token, convert to bf16
      int n = t >> 3, kp = (t & 7) * 4;
      float4 v = *(const float4*)(codebook + (size_t)idxs[n] * CD_DIM + k0 + kp);
      v4bf o = {(__bf16)v.x, (__bf16)v.y, (__bf16)v.z, (__bf16)v.w};
      *(v4bf*)(Bsm + n * 32 + kp) = o;
    }
    __syncthreads();
    v16bf a  = ld_frag_a(Asm, 32, w * 16 + col, half);
    v16bf b0 = ld_frag_b(Bsm, 32, col, half);
    v16bf b1 = ld_frag_b(Bsm, 32, 16 + col, half);
    acc0 = wmma_bf16(a, b0, acc0);
    acc1 = wmma_bf16(a, b1, acc1);
  }
#pragma unroll
  for (int r = 0; r < 8; r++) {
    int m = m0 + w * 16 + r + 8 * half;
    float bias = out_b[m];
    out[(size_t)m * T_DIM + n0 + col]      = acc0[r] + bias;
    out[(size_t)m * T_DIM + n0 + 16 + col] = acc1[r] + bias;
  }
}

extern "C" void kernel_launch(void* const* d_in, const int* in_sizes, int n_in,
                              void* d_out, int out_size, void* d_ws, size_t ws_size,
                              hipStream_t stream) {
  (void)in_sizes; (void)n_in; (void)out_size; (void)ws_size;
  const float* z     = (const float*)d_in[0];
  const float* in_v  = (const float*)d_in[1];
  const float* in_g  = (const float*)d_in[2];
  const float* in_b  = (const float*)d_in[3];
  const float* out_v = (const float*)d_in[4];
  const float* out_g = (const float*)d_in[5];
  const float* out_b = (const float*)d_in[6];
  const float* cb    = (const float*)d_in[7];

  char* ws = (char*)d_ws;
  size_t off = 0;
  auto alloc = [&](size_t bytes) {
    void* p = ws + off;
    off = (off + bytes + 255) & ~(size_t)255;
    return p;
  };
  __bf16* Win  = (__bf16*)alloc((size_t)CD_DIM * D_DIM * 2);   // 512 KB
  __bf16* Wout = (__bf16*)alloc((size_t)D_DIM * CD_DIM * 2);   // 512 KB
  __bf16* cbu  = (__bf16*)alloc((size_t)K_DIM * CD_DIM * 2);   // 4 MB
  __bf16* zb   = (__bf16*)alloc((size_t)D_DIM * T_DIM * 2);    // 32 MB
  __bf16* enc  = (__bf16*)alloc((size_t)T_DIM * CD_DIM * 2);   // 8 MB
  int*    idxb = (int*)alloc((size_t)T_DIM * 4);               // 64 KB

  float* out  = (float*)d_out;
  float* fidx = out + (size_t)D_DIM * T_DIM;

  k_rownorm<<<CD_DIM, 256, 0, stream>>>(in_v, in_g, Win, D_DIM, 0);
  k_rownorm<<<D_DIM, 256, 0, stream>>>(out_v, out_g, Wout, CD_DIM, 0);
  k_rownorm<<<K_DIM, 256, 0, stream>>>(cb, nullptr, cbu, CD_DIM, 1);
  k_f32_to_bf16<<<(D_DIM * T_DIM / 4 + 255) / 256, 256, 0, stream>>>(
      z, zb, (long long)D_DIM * T_DIM / 4);
  k_inproj_enc<<<T_DIM / 32, 256, 0, stream>>>(Win, zb, in_b, enc);
  k_argmax<<<T_DIM / 32, 256, 0, stream>>>(enc, cbu, idxb, fidx);
  dim3 g3(T_DIM / 32, D_DIM / 128);
  k_outproj<<<g3, 256, 0, stream>>>(Wout, cb, idxb, out_b, out);
}
